// MambaBlock_18691697672243
// MI455X (gfx1250) — matrix operand order repaired and checked by reading, not compiled, not run on hardware
//
#include <hip/hip_runtime.h>

// ---------------------------------------------------------------------------
// MambaBlock for MI455X (gfx1250, wave32, WMMA).
// B=8, T=4096, D=2048, S=64.
//   a = tanh(x @ Wa^T + ba); b = tanh(x @ Wb^T + bb)   [proj_kernel, WMMA bf16]
//   h_t = a_t*h_{t-1} + b_t                             [scan_kernel, 2-level scan]
//   out = x + hs @ Wout^T + bout                        [out_kernel, WMMA bf16]
// Weights staged through LDS (double-buffered in proj); all 8 B-fragments of a
// k-step are ds_loaded as one batch so WMMAs overlap LDS latency.
// ---------------------------------------------------------------------------

#define BDIM   8
#define TDIM   4096
#define DDIM   2048
#define SDIM   64
#define MROWS  (BDIM * TDIM)   // 32768
#define KSTEPS (DDIM / 32)     // 64

typedef __attribute__((ext_vector_type(16))) __bf16 v16bf;
typedef __attribute__((ext_vector_type(8)))  __bf16 v8bf;
typedef __attribute__((ext_vector_type(8)))  float  v8f;
typedef __attribute__((ext_vector_type(4)))  float  v4f;

// ---- fragment builders ----------------------------------------------------
// A-matrix (16x32 bf16) per-lane layout (ISA 7.12.2):
//   row M = lane&15; elems e=0..7 hold K = kg+0..7, e=8..15 hold K = kg+16..23
//   with kg = (lane>>4)*8. Source fp32 row-major, converted in-register.
__device__ __forceinline__ v16bf a_frag_from_f32(const float* __restrict__ p) {
    v4f f0 = *(const v4f*)(p);
    v4f f1 = *(const v4f*)(p + 4);
    v4f f2 = *(const v4f*)(p + 16);
    v4f f3 = *(const v4f*)(p + 20);
    v16bf r;
#pragma unroll
    for (int i = 0; i < 4; ++i) {
        r[i]      = (__bf16)f0[i];
        r[4 + i]  = (__bf16)f1[i];
        r[8 + i]  = (__bf16)f2[i];
        r[12 + i] = (__bf16)f3[i];
    }
    return r;
}

// B-matrix (32x16 bf16): N = lane&15, e=0..7 hold K = (lane>>4)*8 + e,
// e=8..15 hold K = 16 + (lane>>4)*8 + (e-8) (2x ds_load_b128 / global_load_b128).
__device__ __forceinline__ v16bf b_frag_load(const __bf16* p) {
    v8bf b0 = *(const v8bf*)(p);
    v8bf b1 = *(const v8bf*)(p + 16);
    v16bf r;
#pragma unroll
    for (int i = 0; i < 8; ++i) { r[i] = b0[i]; r[8 + i] = b1[i]; }
    return r;
}

// Branch-free tanh: exact identity 1 - 2/(exp(2|x|)+1), sign restored.
// One v_exp_f32 + fast rcp; avoids ocml's divergent EXEC-mask blob.
__device__ __forceinline__ float fast_tanh(float v) {
    float ax = __builtin_fabsf(v);
    float e  = __expf(2.0f * ax);
    float t  = 1.0f - __fdividef(2.0f, e + 1.0f);
    return __builtin_copysignf(t, v);
}

// ---- kernel 0: weight fp32 -> bf16 (once; tiny, stays in L2) --------------
__global__ void __launch_bounds__(256)
cvt_kernel(const float* __restrict__ Wa, const float* __restrict__ Wb,
           const float* __restrict__ Wo,
           __bf16* __restrict__ WaB, __bf16* __restrict__ WbB,
           __bf16* __restrict__ WoB) {
    int i = blockIdx.x * 256 + threadIdx.x;       // SDIM*DDIM = 131072 each
    if (i < SDIM * DDIM) {
        WaB[i] = (__bf16)Wa[i];
        WbB[i] = (__bf16)Wb[i];
        WoB[i] = (__bf16)Wo[i];
    }
}

// ---- kernel 1: fused dual projection GEMM + bias + tanh -------------------
// grid.x = MROWS/128; 256 threads = 8 waves; wave owns 16 rows x 64 cols of
// BOTH a and b. Weight k-slab (Wa||Wb : 128 rows x 32 K bf16 = 8KB) is staged
// in double-buffered LDS; iter i computes slab i from buf(i&1) while the
// global load for slab i+2 is in flight and slab i+1 is ds_store'd.
__global__ void __launch_bounds__(256)
proj_kernel(const float* __restrict__ x,
            const __bf16* __restrict__ WaB, const __bf16* __restrict__ WbB,
            const float* __restrict__ ba, const float* __restrict__ bb,
            float* __restrict__ aOut, float* __restrict__ bOut) {
    __shared__ __bf16 sW[2][128 * 32];   // [buf][row][k]; rows 0-63 Wa, 64-127 Wb

    const int tid   = threadIdx.x;
    const int wave  = tid >> 5;
    const int lane  = tid & 31;
    const int nlane = lane & 15;
    const int kg    = (lane >> 4) * 8;    // A-frag K-half
    const int kofs  = (lane >> 4) * 8;    // B-frag K-half
    const int rowBase = blockIdx.x * 128 + wave * 16;

    const float* __restrict__ rowPtr = x + (size_t)(rowBase + nlane) * DDIM;

    // staging assignment: thread stages 16 bf16 (32B): row = tid>>1, k = (tid&1)*16
    const int srow = tid >> 1;            // 0..127
    const int skp  = (tid & 1) * 16;
    const __bf16* __restrict__ wsrc =
        (srow < 64 ? WaB + (size_t)srow * DDIM
                   : WbB + (size_t)(srow - 64) * DDIM) + skp;
    __bf16* const sdst = &sW[0][srow * 32 + skp];

    // prologue: slab0 -> buf0; preload slab1 into regs
    v16bf stage = *(const v16bf*)(wsrc);
    *(v16bf*)(sdst) = stage;
    __syncthreads();
    stage = *(const v16bf*)(wsrc + 32);

    v8f accA[4] = {}, accB[4] = {};

    for (int i = 0; i < KSTEPS; ++i) {
        const int k0  = i * 32;
        const int buf = i & 1;

        // slab i+1 -> other buffer (iter i-1's reads of it are barrier-fenced)
        if (i + 1 < KSTEPS)
            *(v16bf*)(sdst + (buf ^ 1) * (128 * 32)) = stage;

        if (k0 + 256 < DDIM) __builtin_prefetch(rowPtr + k0 + 256, 0, 1);

        // batch all 8 B-fragments (one clause of 16 ds_load_b128), then build
        // the A-fragment while those are in flight, then run the 8 WMMAs.
        const __bf16* sbase = &sW[buf][0];
        v16bf bfrag[8];
#pragma unroll
        for (int n = 0; n < 4; ++n) {
            bfrag[n]     = b_frag_load(sbase + (n * 16 + nlane) * 32 + kofs);
            bfrag[4 + n] = b_frag_load(sbase + (64 + n * 16 + nlane) * 32 + kofs);
        }
        v16bf af = a_frag_from_f32(rowPtr + k0 + kg);
#pragma unroll
        for (int n = 0; n < 4; ++n) {
            accA[n] = __builtin_amdgcn_wmma_f32_16x16x32_bf16(
                false, af, false, bfrag[n], (short)0, accA[n], false, false);
            accB[n] = __builtin_amdgcn_wmma_f32_16x16x32_bf16(
                false, af, false, bfrag[4 + n], (short)0, accB[n], false, false);
        }
        __syncthreads();
        if (i + 2 < KSTEPS)
            stage = *(const v16bf*)(wsrc + (size_t)(i + 2) * 32);
    }

    // Epilogue. C/D layout: row M = 8*(lane>>4)+r, col N = lane&15 (+16*n).
    const int mrow = rowBase + (lane >> 4) * 8;
#pragma unroll
    for (int n = 0; n < 4; ++n) {
        const int   col   = n * 16 + nlane;
        const float biasA = ba[col];
        const float biasB = bb[col];
#pragma unroll
        for (int r = 0; r < 8; ++r) {
            const size_t idx = (size_t)(mrow + r) * SDIM + col;
            aOut[idx] = fast_tanh(accA[n][r] + biasA);
            bOut[idx] = fast_tanh(accB[n][r] + biasB);
        }
    }
}

// ---- kernel 2: affine recurrence, two-level scan --------------------------
// One block per batch b: 512 threads = 8 chunks x 64 states. Pass A computes
// each chunk's affine transform (A,B); compose through LDS; pass B replays
// with the correct h_in. Serial chain ~1030 FMAs instead of 4096; a,b are L2-hot.
__global__ void __launch_bounds__(512)
scan_kernel(const float* __restrict__ aIn, const float* __restrict__ bIn,
            float* __restrict__ hs) {
    __shared__ float sA[512];
    __shared__ float sB[512];
    const int CLEN  = TDIM / 8;            // 512
    const int s     = threadIdx.x & 63;
    const int chunk = threadIdx.x >> 6;
    const size_t base =
        ((size_t)blockIdx.x * TDIM + (size_t)chunk * CLEN) * SDIM + s;

    float A = 1.0f, Bc = 0.0f;
#pragma unroll 4
    for (int t = 0; t < CLEN; ++t) {
        const float at = aIn[base + (size_t)t * SDIM];
        const float bt = bIn[base + (size_t)t * SDIM];
        A  = at * A;
        Bc = fmaf(at, Bc, bt);
    }
    sA[threadIdx.x] = A;
    sB[threadIdx.x] = Bc;
    __syncthreads();

    float h = 0.0f;
    for (int c = 0; c < chunk; ++c)
        h = fmaf(sA[c * 64 + s], h, sB[c * 64 + s]);

#pragma unroll 4
    for (int t = 0; t < CLEN; ++t) {
        const float at = aIn[base + (size_t)t * SDIM];
        const float bt = bIn[base + (size_t)t * SDIM];
        h = fmaf(at, h, bt);
        hs[base + (size_t)t * SDIM] = h;
    }
}

// ---- kernel 3: out = x + hs @ Wout^T + bout -------------------------------
// grid = (MROWS/128, DDIM/64); wave = 16 rows x 64 cols, K = 64 (2 steps).
// Wout tile (64 n-rows x 64 K bf16 = 8KB) staged once in LDS per block.
__global__ void __launch_bounds__(256)
out_kernel(const float* __restrict__ x, const float* __restrict__ hs,
           const __bf16* __restrict__ WoB, const float* __restrict__ bout,
           float* __restrict__ out) {
    __shared__ __bf16 sW[64 * 64];        // [n_local][k]

    const int tid   = threadIdx.x;
    const int wave  = tid >> 5;
    const int lane  = tid & 31;
    const int nlane = lane & 15;
    const int kg    = (lane >> 4) * 8;
    const int kofs  = (lane >> 4) * 8;
    const int rowBase = blockIdx.x * 128 + wave * 16;
    const int colBase = blockIdx.y * 64;

    {   // stage: thread loads 16 bf16: n_local = tid>>2, k = (tid&3)*16
        const int srow = tid >> 2, skp = (tid & 3) * 16;
        *(v16bf*)&sW[srow * 64 + skp] =
            *(const v16bf*)(WoB + (size_t)(colBase + srow) * SDIM + skp);
    }
    __syncthreads();

    const float* __restrict__ hRow = hs + (size_t)(rowBase + nlane) * SDIM;

    v8f acc[4] = {};
#pragma unroll
    for (int k0 = 0; k0 < SDIM; k0 += 32) {
        // batch 4 B-fragments, then A while they are in flight, then WMMAs
        v16bf bfrag[4];
#pragma unroll
        for (int n = 0; n < 4; ++n)
            bfrag[n] = b_frag_load(&sW[(n * 16 + nlane) * 64 + k0 + kofs]);
        v16bf af = a_frag_from_f32(hRow + k0 + kg);
#pragma unroll
        for (int n = 0; n < 4; ++n)
            acc[n] = __builtin_amdgcn_wmma_f32_16x16x32_bf16(
                false, af, false, bfrag[n], (short)0, acc[n], false, false);
    }

    const int mrow = rowBase + (lane >> 4) * 8;
#pragma unroll
    for (int n = 0; n < 4; ++n) {
        const int   d  = colBase + n * 16 + nlane;
        const float bo = bout[d];
#pragma unroll
        for (int r = 0; r < 8; ++r) {
            const size_t idx = (size_t)(mrow + r) * DDIM + d;
            out[idx] = x[idx] + acc[n][r] + bo;
        }
    }
}

// ---------------------------------------------------------------------------
extern "C" void kernel_launch(void* const* d_in, const int* in_sizes, int n_in,
                              void* d_out, int out_size, void* d_ws, size_t ws_size,
                              hipStream_t stream) {
    const float* x    = (const float*)d_in[0];
    const float* Wa   = (const float*)d_in[1];
    const float* ba   = (const float*)d_in[2];
    const float* Wb   = (const float*)d_in[3];
    const float* bb   = (const float*)d_in[4];
    const float* Wout = (const float*)d_in[5];
    const float* bout = (const float*)d_in[6];
    float* out = (float*)d_out;

    // workspace layout
    char* ws = (char*)d_ws;
    float*  aBuf  = (float*)(ws);                              // 8 MB
    float*  bBuf  = (float*)(ws + ((size_t)8  << 20));         // 8 MB
    float*  hsBuf = (float*)(ws + ((size_t)16 << 20));         // 8 MB
    __bf16* WaB   = (__bf16*)(ws + ((size_t)24 << 20));        // 256 KB
    __bf16* WbB   = WaB + (size_t)SDIM * DDIM;
    __bf16* WoB   = WbB + (size_t)SDIM * DDIM;

    // 0) weights -> bf16
    cvt_kernel<<<(SDIM * DDIM + 255) / 256, 256, 0, stream>>>(Wa, Wb, Wout,
                                                              WaB, WbB, WoB);
    // 1) fused dual projection (WMMA bf16, fp32 accumulate) + bias + tanh
    proj_kernel<<<MROWS / 128, 256, 0, stream>>>(x, WaB, WbB, ba, bb, aBuf, bBuf);
    // 2) affine recurrence (two-level scan)
    scan_kernel<<<BDIM, 512, 0, stream>>>(aBuf, bBuf, hsBuf);
    // 3) output projection + residual + bias
    dim3 grid3(MROWS / 128, DDIM / 64);
    out_kernel<<<grid3, 256, 0, stream>>>(x, hsBuf, WoB, bout, out);
}
